// ConditionalGNN_91001767068103
// MI455X (gfx1250) — matrix operand order repaired and checked by reading, not compiled, not run on hardware
//
#include <hip/hip_runtime.h>
#include <hip/hip_bf16.h>

typedef __attribute__((ext_vector_type(16))) __bf16 v16bf;
typedef __attribute__((ext_vector_type(8)))  float  v8f;

#define N_NODES 50000
#define N_EDGES 800000
#define NODE_IN 32
#define EDGE_IN 16
#define HID 128
#define N_GRAPHS 256
#define T_EMBD 64
#define BN_EPS 1e-5f

#define WMMA_BF16(A, B, C) \
  __builtin_amdgcn_wmma_f32_16x16x32_bf16(false, (A), false, (B), (short)0, (C), false, false)

// ---------------------------------------------------------------------------
// WMMA operand layouts (ISA 7.12.2, wave32):
//  A 16x32 bf16: lane holds row m = lane&15, half = lane>>4.
//    packed elems i=0..15 -> K = 16*half + i (this file's order; ISA table order differs)
//  B 32x16 bf16 (KxN): lane holds col n = lane&15; lanes 0-15 K=kb..kb+15,
//    lanes 16-31 K=kb+16..kb+31 (two bf16 per VGPR); same order as A, not the ISA table's.
//  C/D 16x16 f32: VGPR r holds (m = r + 8*half, n = lane&15).
// B weights are PRE-SWIZZLED so each lane's 16 bf16 are contiguous (32 B):
//    swz[(((c*8 + t)*32) + lane)*16 + i]  ->  two global_load_b128 per tile.
// ---------------------------------------------------------------------------
__device__ __forceinline__ int a_k(int i, int half) {
  return 16 * half + i;
}

__device__ __forceinline__ v16bf load_b_swz(const __bf16* __restrict__ Wsw,
                                            int c, int t, int lane) {
  return *(const v16bf*)(Wsw + ((((c << 3) + t) << 5) + lane) * 16);
}

// ------------------------------- utility -----------------------------------
__global__ void zero_f32(float* __restrict__ p, int n) {
  int i = blockIdx.x * blockDim.x + threadIdx.x;
  if (i < n) p[i] = 0.0f;
}

// Convert + swizzle a KxN(=128) fp32 weight into WMMA lane order, zero-padding
// rows k >= kmax (used to pad the edge weight K:16 -> 32).
__global__ void swz_b(const float* __restrict__ src, __bf16* __restrict__ dst,
                      int K, int kmax) {
  int o = blockIdx.x * blockDim.x + threadIdx.x;
  if (o >= K * HID) return;
  int i = o & 15, l = (o >> 4) & 31, t = (o >> 9) & 7, c = o >> 12;
  int k = c * 32 + 16 * (l >> 4) + i;
  int n = t * 16 + (l & 15);
  dst[o] = (k < kmax) ? (__bf16)src[k * HID + n] : (__bf16)0.0f;
}

// --------------------- node encoder: h = x @ node_w + b ---------------------
__global__ __launch_bounds__(256) void node_enc(const float* __restrict__ x,
                                                const __bf16* __restrict__ wb,
                                                const float* __restrict__ bias,
                                                float* __restrict__ h) {
  int wave = (int)((blockIdx.x * blockDim.x + threadIdx.x) >> 5);
  int lane = threadIdx.x & 31;
  if (wave >= N_NODES / 16) return;
  int row0 = wave * 16, half = lane >> 4, nl = lane & 15;
  int m = row0 + nl;
  v16bf a;
#pragma unroll
  for (int i = 0; i < 16; ++i) a[i] = (__bf16)x[m * NODE_IN + a_k(i, half)];
#pragma unroll
  for (int t = 0; t < 8; ++t) {
    v8f acc = {};
    acc = WMMA_BF16(a, load_b_swz(wb, 0, t, lane), acc);
    int n = t * 16 + nl;
    float bv = bias[n];
#pragma unroll
    for (int r = 0; r < 8; ++r) h[(row0 + 8 * half + r) * HID + n] = acc[r] + bv;
  }
}

// --- fused GINE edge kernel: e = eattr@W+b; msg = relu(h[src]+e); agg[dst]+= -
__global__ __launch_bounds__(256) void edge_gine(const float* __restrict__ eattr,
                                                 const int* __restrict__ eidx,
                                                 const __bf16* __restrict__ wbE,
                                                 const float* __restrict__ eb,
                                                 const float* __restrict__ h,
                                                 float* __restrict__ agg) {
  int wave = (int)((blockIdx.x * blockDim.x + threadIdx.x) >> 5);
  int lane = threadIdx.x & 31;
  if (wave >= N_EDGES / 16) return;
  int e0 = wave * 16, half = lane >> 4, nl = lane & 15;
  int myedge = e0 + nl;
  int msrc = eidx[myedge];            // edge_index[0][e]
  int mdst = eidx[N_EDGES + myedge];  // edge_index[1][e]
  v16bf a;                            // 16 edges x (K=16 real, 16 zero-pad)
#pragma unroll
  for (int i = 0; i < 16; ++i) {
    int k = a_k(i, half);
    a[i] = (k < EDGE_IN) ? (__bf16)eattr[myedge * EDGE_IN + k] : (__bf16)0.0f;
  }
#pragma unroll
  for (int t = 0; t < 8; ++t) {
    v8f acc = {};
    acc = WMMA_BF16(a, load_b_swz(wbE, 0, t, lane), acc);
    int n = t * 16 + nl;
    float bv = eb[n];
#pragma unroll
    for (int r = 0; r < 8; ++r) {
      int m = 8 * half + r;  // edge within tile
      int s = __shfl(msrc, m, 32);
      int d = __shfl(mdst, m, 32);
      float msg = fmaxf(acc[r] + bv + h[s * HID + n], 0.0f);  // L2-resident gather
      atomicAdd(&agg[d * HID + n], msg);                       // L2-resident scatter
    }
  }
}

// --- fused node MLP + BN statistics.  zin=(1+eps)*h+agg; u=relu(zin@W1+b1);
//     z=u@W2+b2 (written over agg); column sum/sumsq accumulated for BN. -----
__global__ __launch_bounds__(128) void mlp_bn_stats(
    const float* __restrict__ h, float* __restrict__ agg, const float* __restrict__ epsArr,
    int layer, const __bf16* __restrict__ w1, const float* __restrict__ b1,
    const __bf16* __restrict__ w2, const float* __restrict__ b2,
    float* __restrict__ colsum, float* __restrict__ colsumsq) {
  __shared__ float ldsu[4][16 * HID];  // per-wave 16x128 ReLU activations
  int wib  = threadIdx.x >> 5;
  int wave = (int)((blockIdx.x * blockDim.x + threadIdx.x) >> 5);
  int lane = threadIdx.x & 31;
  if (wave >= N_NODES / 16) return;
  int row0 = wave * 16, half = lane >> 4, nl = lane & 15;
  float epsv = 1.0f + epsArr[layer];
  int m = row0 + nl;

  v16bf A[4];  // zin tile 16x128 as 4 K-chunks, built in registers
#pragma unroll
  for (int c = 0; c < 4; ++c)
#pragma unroll
    for (int i = 0; i < 16; ++i) {
      int k = c * 32 + a_k(i, half);
      A[c][i] = (__bf16)(epsv * h[m * HID + k] + agg[m * HID + k]);
    }

  float* myu = ldsu[wib];
#pragma unroll
  for (int t = 0; t < 8; ++t) {
    v8f acc = {};
#pragma unroll
    for (int c = 0; c < 4; ++c) acc = WMMA_BF16(A[c], load_b_swz(w1, c, t, lane), acc);
    int n = t * 16 + nl;
    float bv = b1[n];
#pragma unroll
    for (int r = 0; r < 8; ++r) myu[(8 * half + r) * HID + n] = fmaxf(acc[r] + bv, 0.0f);
  }

  v16bf A2[4];  // re-tile u from LDS (same-wave DS ops are in-order)
#pragma unroll
  for (int c = 0; c < 4; ++c)
#pragma unroll
    for (int i = 0; i < 16; ++i) A2[c][i] = (__bf16)myu[nl * HID + c * 32 + a_k(i, half)];

#pragma unroll
  for (int t = 0; t < 8; ++t) {
    v8f acc = {};
#pragma unroll
    for (int c = 0; c < 4; ++c) acc = WMMA_BF16(A2[c], load_b_swz(w2, c, t, lane), acc);
    int n = t * 16 + nl;
    float bv = b2[n], ls = 0.0f, lq = 0.0f;
#pragma unroll
    for (int r = 0; r < 8; ++r) {
      float z = acc[r] + bv;
      agg[(row0 + 8 * half + r) * HID + n] = z;  // rows fully consumed above: safe reuse
      ls += z;
      lq += z * z;
    }
    atomicAdd(&colsum[n], ls);
    atomicAdd(&colsumsq[n], lq);
  }
}

// ---------------- BN normalize (+ReLU): h = relu(bn(z)) ---------------------
__global__ void bn_relu(const float* __restrict__ z, const float* __restrict__ cs,
                        const float* __restrict__ cq, const float* __restrict__ gamma,
                        const float* __restrict__ beta, float* __restrict__ h) {
  int idx = blockIdx.x * blockDim.x + threadIdx.x;
  if (idx >= N_NODES * HID) return;
  int n = idx & (HID - 1);
  float mean = cs[n] * (1.0f / N_NODES);
  float var  = cq[n] * (1.0f / N_NODES) - mean * mean;
  float v = (z[idx] - mean) * rsqrtf(var + BN_EPS) * gamma[n] + beta[n];
  h[idx] = fmaxf(v, 0.0f);
}

// ---------------- global add pool ------------------------------------------
__global__ void pool_add(const float* __restrict__ h, const int* __restrict__ batch,
                         float* __restrict__ g) {
  int idx = blockIdx.x * blockDim.x + threadIdx.x;
  if (idx >= N_NODES * HID) return;
  int node = idx >> 7, c = idx & (HID - 1);
  atomicAdd(&g[batch[node] * HID + c], h[idx]);
}

// ---------------- head: 256 graphs, tiny MLP (VALU) -------------------------
__global__ __launch_bounds__(64) void head(const float* __restrict__ g, const int* __restrict__ tgt,
                                           const float* __restrict__ temb,
                                           const float* __restrict__ w1, const float* __restrict__ b1,
                                           const float* __restrict__ w2, const float* __restrict__ b2,
                                           const float* __restrict__ w3, const float* __restrict__ b3,
                                           float* __restrict__ out) {
  __shared__ float cat[HID + T_EMBD];
  __shared__ float u1[HID];
  __shared__ float u2[HID / 2];
  int gph = blockIdx.x, t = threadIdx.x;
  int ti = tgt[gph];
  cat[t] = g[gph * HID + t];
  cat[t + 64] = g[gph * HID + t + 64];
  cat[t + 128] = temb[ti * T_EMBD + t];
  __syncthreads();
  for (int c = t; c < HID; c += 64) {
    float s = b1[c];
    for (int k = 0; k < HID + T_EMBD; ++k) s += cat[k] * w1[k * HID + c];
    u1[c] = fmaxf(s, 0.0f);
  }
  __syncthreads();
  {
    float s = b2[t];
    for (int k = 0; k < HID; ++k) s += u1[k] * w2[k * 64 + t];
    u2[t] = fmaxf(s, 0.0f);
  }
  __syncthreads();
  if (t == 0) {
    float s = b3[0];
    for (int k = 0; k < 64; ++k) s += u2[k] * w3[k];
    out[gph] = s;
  }
}

// ---------------------------------------------------------------------------
extern "C" void kernel_launch(void* const* d_in, const int* in_sizes, int n_in,
                              void* d_out, int out_size, void* d_ws, size_t ws_size,
                              hipStream_t stream) {
  (void)in_sizes; (void)n_in; (void)out_size; (void)ws_size;
  const float* x      = (const float*)d_in[0];
  const int*   eidx   = (const int*)  d_in[1];
  const float* eattr  = (const float*)d_in[2];
  const int*   batch  = (const int*)  d_in[3];
  const int*   tgt    = (const int*)  d_in[4];
  const float* node_w = (const float*)d_in[5];
  const float* node_b = (const float*)d_in[6];
  const float* epsA   = (const float*)d_in[7];
  const float* edge_w = (const float*)d_in[8];
  const float* edge_b = (const float*)d_in[9];
  const float* mlp_w1 = (const float*)d_in[10];
  const float* mlp_b1 = (const float*)d_in[11];
  const float* mlp_w2 = (const float*)d_in[12];
  const float* mlp_b2 = (const float*)d_in[13];
  const float* bn_g   = (const float*)d_in[14];
  const float* bn_b   = (const float*)d_in[15];
  const float* temb   = (const float*)d_in[16];
  const float* h1_w   = (const float*)d_in[17];
  const float* h1_b   = (const float*)d_in[18];
  const float* h2_w   = (const float*)d_in[19];
  const float* h2_b   = (const float*)d_in[20];
  const float* h3_w   = (const float*)d_in[21];
  const float* h3_b   = (const float*)d_in[22];
  float* out = (float*)d_out;

  // workspace layout (bytes)
  char* ws = (char*)d_ws;
  float*  h   = (float*)(ws + 0);                    // 50000*128*4 = 25,600,000
  float*  z   = (float*)(ws + 25600000);             // agg, then z (in-place)
  float*  cs  = (float*)(ws + 51200000);             // 128 col sums
  float*  cq  = (float*)(ws + 51200512);             // 128 col sumsq (contiguous)
  float*  g   = (float*)(ws + 51201024);             // 256*128
  __bf16* wbn = (__bf16*)(ws + 51332096);            // 32x128 swizzled
  __bf16* wbe = (__bf16*)(ws + 51340288);            // 3 x 32x128 swizzled (K padded)
  __bf16* wb1 = (__bf16*)(ws + 51364864);            // 3 x 128x128 swizzled
  __bf16* wb2 = (__bf16*)(ws + 51463168);            // 3 x 128x128 swizzled (end 51,561,472)

  // weight convert + WMMA-order swizzle (cheap, every call -> deterministic)
  swz_b<<<16, 256, 0, stream>>>(node_w, wbn, 32, 32);
  for (int l = 0; l < 3; ++l) {
    swz_b<<<16, 256, 0, stream>>>(edge_w + l * EDGE_IN * HID, wbe + l * 32 * HID, 32, EDGE_IN);
    swz_b<<<64, 256, 0, stream>>>(mlp_w1 + l * HID * HID, wb1 + l * HID * HID, HID, HID);
    swz_b<<<64, 256, 0, stream>>>(mlp_w2 + l * HID * HID, wb2 + l * HID * HID, HID, HID);
  }

  node_enc<<<391, 256, 0, stream>>>(x, wbn, node_b, h);  // 3125 waves

  for (int l = 0; l < 3; ++l) {
    zero_f32<<<25000, 256, 0, stream>>>(z, N_NODES * HID);
    zero_f32<<<1, 256, 0, stream>>>(cs, 256);  // cs+cq contiguous
    edge_gine<<<6250, 256, 0, stream>>>(eattr, eidx, wbe + l * 32 * HID,
                                        edge_b + l * HID, h, z);  // 50000 waves
    mlp_bn_stats<<<782, 128, 0, stream>>>(h, z, epsA, l, wb1 + l * HID * HID,
                                          mlp_b1 + l * HID, wb2 + l * HID * HID,
                                          mlp_b2 + l * HID, cs, cq);
    bn_relu<<<25000, 256, 0, stream>>>(z, cs, cq, bn_g + l * HID, bn_b + l * HID, h);
  }

  zero_f32<<<128, 256, 0, stream>>>(g, N_GRAPHS * HID);
  pool_add<<<25000, 256, 0, stream>>>(h, batch, g);
  head<<<N_GRAPHS, 64, 0, stream>>>(g, tgt, temb, h1_w, h1_b, h2_w, h2_b, h3_w, h3_b, out);
}
